// Int8Linear_45689862095007
// MI455X (gfx1250) — 8-bit matrix operand order repaired and checked by reading, compiled, not run on hardware
//
#include <hip/hip_runtime.h>
#include <hip/hip_bf16.h>
#include <stdint.h>


#define T_TOKENS 8192
#define K_FEAT   4096
#define N_FEAT   4096

typedef int v8i __attribute__((ext_vector_type(8)));

#if defined(__AMDGCN__) && \
    __has_builtin(__builtin_amdgcn_global_load_async_to_lds_b128) && \
    __has_builtin(__builtin_amdgcn_s_wait_asynccnt)
#define USE_ASYNC_LDS 1
typedef int vec4i __attribute__((vector_size(16)));
typedef __attribute__((address_space(1))) vec4i* async_gptr;
typedef __attribute__((address_space(3))) vec4i* async_lptr;
#else
#define USE_ASYNC_LDS 0
#endif

__global__ __launch_bounds__(256)
void quant_rows_kernel(const float* __restrict__ x,
                       signed char* __restrict__ qx,
                       float* __restrict__ xscale) {
    const int t    = blockIdx.x;
    const int tid  = threadIdx.x;
    const int lane = tid & 31;
    const int wid  = tid >> 5;

    const float* row = x + (size_t)t * K_FEAT;
    const float4* rv = (const float4*)row;
    float4 f0 = rv[tid * 4 + 0];
    float4 f1 = rv[tid * 4 + 1];
    float4 f2 = rv[tid * 4 + 2];
    float4 f3 = rv[tid * 4 + 3];

    float vals[16] = { f0.x, f0.y, f0.z, f0.w, f1.x, f1.y, f1.z, f1.w,
                       f2.x, f2.y, f2.z, f2.w, f3.x, f3.y, f3.z, f3.w };

    float m = 0.0f;
    #pragma unroll
    for (int i = 0; i < 16; ++i) m = fmaxf(m, fabsf(vals[i]));

    #pragma unroll
    for (int off = 16; off > 0; off >>= 1)
        m = fmaxf(m, __shfl_xor(m, off, 32));

    __shared__ float smax[8];
    if (lane == 0) smax[wid] = m;
    __syncthreads();
    float amax = smax[0];
    #pragma unroll
    for (int i = 1; i < 8; ++i) amax = fmaxf(amax, smax[i]);

    const float inv = (amax > 0.0f) ? (127.0f / amax) : 0.0f;

    int q[16];
    #pragma unroll
    for (int i = 0; i < 16; ++i) {
        int v = (int)rintf(vals[i] * inv);
        v = v > 127 ? 127 : (v < -127 ? -127 : v);
        q[i] = v;
    }

    int4 packed;
    packed.x = (q[0]  & 255) | ((q[1]  & 255) << 8) | ((q[2]  & 255) << 16) | (q[3]  << 24);
    packed.y = (q[4]  & 255) | ((q[5]  & 255) << 8) | ((q[6]  & 255) << 16) | (q[7]  << 24);
    packed.z = (q[8]  & 255) | ((q[9]  & 255) << 8) | ((q[10] & 255) << 16) | (q[11] << 24);
    packed.w = (q[12] & 255) | ((q[13] & 255) << 8) | ((q[14] & 255) << 16) | (q[15] << 24);

    ((int4*)(qx + (size_t)t * K_FEAT))[tid] = packed;

    if (tid == 0) xscale[t] = amax * (1.0f / 127.0f);
}

#define LDA 80

__device__ __forceinline__
void stage_tile(const signed char* __restrict__ gA,
                const signed char* __restrict__ gB,
                signed char* sA, signed char* sB, int tid) {
    #pragma unroll
    for (int j = 0; j < 2; ++j) {
        const int c   = tid + 256 * j;
        const int row = c >> 2;
        const int kc  = (c & 3) * 16;
        const size_t goff = (size_t)row * K_FEAT + kc;
        const int    loff = row * LDA + kc;
#if USE_ASYNC_LDS
        __builtin_amdgcn_global_load_async_to_lds_b128(
            (async_gptr)(gA + goff), (async_lptr)(sA + loff), 0, 0);
        __builtin_amdgcn_global_load_async_to_lds_b128(
            (async_gptr)(gB + goff), (async_lptr)(sB + loff), 0, 0);
#else
        *(int4*)&sA[loff] = *(const int4*)&gA[goff];
        *(int4*)&sB[loff] = *(const int4*)&gB[goff];
#endif
    }
}

__device__ __forceinline__ void tile_sync() {
#if USE_ASYNC_LDS
    __builtin_amdgcn_s_wait_asynccnt(0);
#endif
    __syncthreads();
}

__device__ __forceinline__ v8i load_b_frag(const signed char* bp) {
    int4 q0 = *(const int4*)(bp +  0);
    int4 q1 = *(const int4*)(bp + 32);
    v8i b;
    b[0] = q0.x; b[1] = q0.y; b[2] = q0.z; b[3] = q0.w;
    b[4] = q1.x; b[5] = q1.y; b[6] = q1.z; b[7] = q1.w;
    return b;
}

__global__ __launch_bounds__(256)
void int8_gemm_wmma_kernel(const signed char* __restrict__ qx,
                           const signed char* __restrict__ wgt,
                           const float* __restrict__ xscale,
                           const float* __restrict__ wscale,
                           const float* __restrict__ bias,
                           float* __restrict__ out) {
    __shared__ signed char lds_a[2][128 * LDA];
    __shared__ signed char lds_b[2][128 * LDA];

    const int tid  = threadIdx.x;
    const int lane = tid & 31;
    const int wid  = tid >> 5;
    const int bm   = blockIdx.y * 128;
    const int bn   = blockIdx.x * 128;

    const signed char* gA = qx  + (size_t)bm * K_FEAT;
    const signed char* gB = wgt + (size_t)bn * K_FEAT;

    v8i acc[8] = {};

    const int a_off = (wid * 16 + (lane & 15)) * LDA + ((lane < 16) ? 0 : 16);
    const int b_off = (lane & 15) * LDA + ((lane < 16) ? 0 : 16);

    stage_tile(gA, gB, lds_a[0], lds_b[0], tid);
    tile_sync();

    const int NSTEP = K_FEAT / 64;
    #pragma unroll 2
    for (int ks = 0; ks < NSTEP; ++ks) {
        const int p = ks & 1;

        if (ks + 1 < NSTEP) {
            const size_t knext = (size_t)(ks + 1) * 64;
            stage_tile(gA + knext, gB + knext, lds_a[p ^ 1], lds_b[p ^ 1], tid);
        }

        v8i a;
        {
            const signed char* ap = &lds_a[p][a_off];
            int2 p0 = *(const int2*)(ap +  0);
            int2 p1 = *(const int2*)(ap +  8);
            int2 p2 = *(const int2*)(ap + 32);
            int2 p3 = *(const int2*)(ap + 40);
            a[0] = p0.x; a[1] = p0.y; a[2] = p1.x; a[3] = p1.y;
            a[4] = p2.x; a[5] = p2.y; a[6] = p3.x; a[7] = p3.y;
        }

        v8i bcur = load_b_frag(&lds_b[p][b_off]);
        #pragma unroll
        for (int nf = 0; nf < 8; ++nf) {
            v8i bnext = (nf < 7) ? load_b_frag(&lds_b[p][b_off + (nf + 1) * 16 * LDA])
                                 : bcur;
            acc[nf] = __builtin_amdgcn_wmma_i32_16x16x64_iu8(
                          true, a, true, bcur, acc[nf], false, false);
            bcur = bnext;
        }

        tile_sync();
    }

    const int m0 = bm + wid * 16 + ((lane < 16) ? 0 : 8);
    const int nc = bn + (lane & 15);

    float xs[8];
    #pragma unroll
    for (int r = 0; r < 8; ++r) xs[r] = xscale[m0 + r];

    #pragma unroll
    for (int nf = 0; nf < 8; ++nf) {
        const int   n  = nc + nf * 16;
        const float ws = wscale[n];
        const float bi = bias[n];
        #pragma unroll
        for (int r = 0; r < 8; ++r) {
            out[(size_t)(m0 + r) * N_FEAT + n] = (float)acc[nf][r] * ws * xs[r] + bi;
        }
    }
}

extern "C" void kernel_launch(void* const* d_in, const int* in_sizes, int n_in,
                              void* d_out, int out_size, void* d_ws, size_t ws_size,
                              hipStream_t stream) {
    (void)in_sizes; (void)n_in; (void)out_size; (void)ws_size;

    const float*       x      = (const float*)d_in[0];
    const signed char* wgt    = (const signed char*)d_in[1];
    const float*       wscale = (const float*)d_in[2];
    const float*       bias   = (const float*)d_in[3];
    float*             out    = (float*)d_out;

    signed char* qx     = (signed char*)d_ws;
    float*       xscale = (float*)((char*)d_ws + (size_t)T_TOKENS * K_FEAT);

    quant_rows_kernel<<<T_TOKENS, 256, 0, stream>>>(x, qx, xscale);

    dim3 grid(N_FEAT / 128, T_TOKENS / 128);
    int8_gemm_wmma_kernel<<<grid, 256, 0, stream>>>(qx, wgt, xscale, wscale, bias, out);
}
